// SAMGuidedGlobalMatcher_48318382080479
// MI455X (gfx1250) — matrix operand order repaired and checked by reading, not compiled, not run on hardware
//
#include <hip/hip_runtime.h>
#include <hip/hip_bf16.h>
#include <math.h>

// ---------------- CDNA5 WMMA types ----------------
typedef __attribute__((ext_vector_type(16))) __bf16 v16bf;
typedef __attribute__((ext_vector_type(8)))  __bf16 v8bf;
typedef __attribute__((ext_vector_type(8)))  float  v8f;
typedef __attribute__((ext_vector_type(8)))  float  vf8;

#define N_TOK   4096
#define CDIM    128
#define TD      192
#define NHEAD   4
#define DHEAD   48
#define DPAD    64          // per-head K padded to 64 for 2x K=32 WMMA
#define TOPK_K  256
#define SCALE_QK 0.14433756729740643f          // 1/sqrt(48)
#define NEG_LOG_DIV (-9.210340371976184f / 96.0f) // -ln(10000)/(token_dim/2)

#define SHUF16(lo, hi) __builtin_shufflevector(lo, hi, 0, 1, 2, 3, 4, 5, 6, 7, \
                                               8, 9, 10, 11, 12, 13, 14, 15)

static __device__ __forceinline__ v8f wmma_bf16(v16bf a, v16bf b, v8f c) {
    return __builtin_amdgcn_wmma_f32_16x16x32_bf16(false, a, false, b, (short)0, c,
                                                   false, false);
}

// IEEE-monotone uint mapping: u0 > u1  <=>  f0 > f1
static __device__ __forceinline__ unsigned mono_u32(float f) {
    unsigned u = __float_as_uint(f);
    return (u & 0x80000000u) ? ~u : (u | 0x80000000u);
}

static __device__ __forceinline__ float pe_val(int n, int t) {
    if (t < 96) {
        float x = (float)(n & 63);
        return __sinf(x * __expf((float)t * NEG_LOG_DIV));
    } else {
        float y = (float)(n >> 6);
        return __cosf(y * __expf((float)(t - 96) * NEG_LOG_DIV));
    }
}

// gfx1250 async global->LDS copy of 16 bytes per lane (ASYNCcnt-tracked).
// VDST holds the LDS byte address = low 32 bits of the generic pointer
// (flat->LDS mapping truncates to addr[31:0]).
static __device__ __forceinline__ void async_g2l_b128(void* lds, const void* gmem) {
    unsigned       l = (unsigned)(unsigned long long)lds;
    unsigned long long g = (unsigned long long)gmem;
    asm volatile("global_load_async_to_lds_b128 %0, %1, off"
                 :: "v"(l), "v"(g) : "memory");
}
static __device__ __forceinline__ void wait_async0() {
    asm volatile("s_wait_asynccnt 0x0" ::: "memory");
}

// ---------------- zero-init (bf16 pad regions) ----------------
__global__ void k_zero(unsigned int* p, int n) {
    for (int i = blockIdx.x * blockDim.x + threadIdx.x; i < n;
         i += gridDim.x * blockDim.x)
        p[i] = 0u;
}

// ---------------- boundary conv: 1->32 3x3 (pad1) -> GELU -> 32->4 1x1 ----------------
__global__ void k_conv(const float* __restrict__ bmap,
                       const float* __restrict__ c1w, const float* __restrict__ c1b,
                       const float* __restrict__ c2w, const float* __restrict__ c2b,
                       float* __restrict__ bfl) {
    int p = blockIdx.x * blockDim.x + threadIdx.x;
    if (p >= N_TOK) return;
    int y = p >> 6, x = p & 63;
    float a0 = 0.f, a1 = 0.f, a2 = 0.f, a3 = 0.f;
    for (int oc = 0; oc < 32; ++oc) {
        float s = c1b[oc];
#pragma unroll
        for (int ky = 0; ky < 3; ++ky) {
            int yy = y + ky - 1;
            if (yy < 0 || yy >= 64) continue;
#pragma unroll
            for (int kx = 0; kx < 3; ++kx) {
                int xx = x + kx - 1;
                if (xx < 0 || xx >= 64) continue;
                s += c1w[oc * 9 + ky * 3 + kx] * bmap[yy * 64 + xx];
            }
        }
        float g = 0.5f * s * (1.f + erff(s * 0.70710678118654752f)); // exact GELU
        a0 += c2w[0 * 32 + oc] * g;
        a1 += c2w[1 * 32 + oc] * g;
        a2 += c2w[2 * 32 + oc] * g;
        a3 += c2w[3 * 32 + oc] * g;
    }
    bfl[0 * N_TOK + p] = a0 + c2b[0];
    bfl[1 * N_TOK + p] = a1 + c2b[1];
    bfl[2 * N_TOK + p] = a2 + c2b[2];
    bfl[3 * N_TOK + p] = a3 + c2b[3];
}

// ---------------- fused QKV projection + bias + PE ----------------
// grid (256 row-tiles, 3 ct-groups, 3 matrices), block 128 (4 waves, wave = col-tile)
// feat layout: (C, N);  W: (128, 192) row-major
// Q,K stored (h, N, 64) bf16 padded; V stored TRANSPOSED (h, 48, N) bf16.
__global__ void k_qkv(const float* __restrict__ feat1, const float* __restrict__ feat2,
                      const float* __restrict__ Wq, const float* __restrict__ bq,
                      const float* __restrict__ Wk, const float* __restrict__ bk,
                      const float* __restrict__ Wv, const float* __restrict__ bv,
                      __bf16* __restrict__ qb, __bf16* __restrict__ kbuf,
                      __bf16* __restrict__ vt) {
    int lane = threadIdx.x & 31;
    int wave = threadIdx.x >> 5;
    int rt = blockIdx.x;
    int ct = blockIdx.y * 4 + wave;
    int z  = blockIdx.z;                 // 0=Q 1=K 2=V
    const float* F = (z == 0) ? feat1 : feat2;
    const float* W = (z == 0) ? Wq : (z == 1 ? Wk : Wv);
    const float* B = (z == 0) ? bq : (z == 1 ? bk : bv);

    int mg   = (lane >> 4) << 3;   // C/D row base: 0 or 8
    int kb8  = mg;                 // unused here (A k base is kb16)
    int kb16 = (lane >> 4) << 4;   // B k base: 0 or 16
    int rowg = rt * 16 + (lane & 15);
    int colg = ct * 16 + (lane & 15);

    v8f acc = {};
    for (int kt = 0; kt < 4; ++kt) {     // K=128 = 4 x 32
        int k0 = kt * 32;
        v16bf a, b;
#pragma unroll
        for (int e = 0; e < 16; ++e) {
            int ka = k0 + kb16 + e;
            a[e] = (__bf16)F[ka * N_TOK + rowg];           // A[row][k] = feat[k][row]
            b[e] = (__bf16)W[(k0 + kb16 + e) * TD + colg]; // B[k][col]
        }
        acc = wmma_bf16(a, b, acc);
    }

    int head = colg / DHEAD;
    int dc   = colg % DHEAD;
#pragma unroll
    for (int r = 0; r < 8; ++r) {
        int m  = r + mg;
        int ng = rt * 16 + m;
        float v = acc[r] + B[colg];
        if (z < 2) v += pe_val(ng, colg);
        if (z == 0)      qb  [((head << 12) + ng) * DPAD + dc] = (__bf16)v;
        else if (z == 1) kbuf[((head << 12) + ng) * DPAD + dc] = (__bf16)v;
        else             vt  [(size_t)(head * DHEAD + dc) * N_TOK + ng] = (__bf16)v;
    }
}

// ---------------- fused scores + top-k softmax + attn*V ----------------
// grid (256 q-tiles, 4 heads), block 128 (4 waves), dynamic LDS = 16*4096 f32 + 48
__global__ void k_attn(const __bf16* __restrict__ qb, const __bf16* __restrict__ kbuf,
                       const __bf16* __restrict__ vt, const float* __restrict__ bfl,
                       const float* __restrict__ segaff,
                       const float* __restrict__ affscale, float* __restrict__ O) {
    extern __shared__ float S[];                 // [16][4096] score/weight strip
    __shared__ float seg_stage[4 * 512];         // per-wave double-buffered 16x16 tiles
    float*    rowmax = &S[16 * N_TOK];
    unsigned* tauu   = (unsigned*)&S[16 * N_TOK + 16];
    float*    rowsum = &S[16 * N_TOK + 32];

    int tid  = threadIdx.x;
    int lane = tid & 31;
    int wave = tid >> 5;
    int qt = blockIdx.x;
    int h  = blockIdx.y;

    int mg   = (lane >> 4) << 3;
    int kb8  = mg;
    int kb16 = (lane >> 4) << 4;
    float affs = affscale[h];
    float* mystage = &seg_stage[wave * 512];

    // ---- phase 1: scores strip (16 x 4096) into LDS ----
    // cache Q A-operands once (K=0..31 and 32..63, padded cols are zero)
    v16bf aq0, aq1;
    {
        const v8bf* qp = (const v8bf*)(qb + ((h << 12) + qt * 16 + (lane & 15)) * DPAD);
        int ca = lane >> 4;                        // kb8/8
        aq0 = SHUF16(qp[ca], qp[ca + 2]);
        aq1 = SHUF16(qp[ca + 4], qp[ca + 6]);
    }
    float bq_r[8];
#pragma unroll
    for (int r = 0; r < 8; ++r) bq_r[r] = bfl[(h << 12) + qt * 16 + r + mg];

    // async-stage first segaff tile (2 x b128 per lane-group: 16 rows x 16 cols f32)
    {
        int row = lane >> 1, ch = lane & 1;
        const float* g = segaff + (size_t)(qt * 16 + row) * N_TOK + wave * 16 + ch * 4;
        float* l = mystage + row * 16 + ch * 4;
        async_g2l_b128(l, g);
        async_g2l_b128(l + 8, g + 8);
    }

    int cur = 0;
    for (int i = 0; i < 64; ++i) {
        int ct   = wave + i * 4;
        int ncol = ct * 16 + (lane & 15);

        wait_async0();                              // current tile resident
        if (i + 1 < 64) {                           // prefetch next tile
            int ctn = ct + 4;
            int row = lane >> 1, ch = lane & 1;
            const float* g = segaff + (size_t)(qt * 16 + row) * N_TOK + ctn * 16 + ch * 4;
            float* l = mystage + (cur ^ 1) * 256 + row * 16 + ch * 4;
            async_g2l_b128(l, g);
            async_g2l_b128(l + 8, g + 8);
        }

        const v8bf* kp = (const v8bf*)(kbuf + ((h << 12) + ncol) * DPAD);
        int cb = lane >> 4;                         // kb8/8
        v16bf b0 = SHUF16(kp[cb], kp[cb + 2]);
        v16bf b1 = SHUF16(kp[cb + 4], kp[cb + 6]);
        v8f acc = {};
        acc = wmma_bf16(aq0, b0, acc);
        acc = wmma_bf16(aq1, b1, acc);

        float bk_n = bfl[(h << 12) + ncol];
        const float* st = mystage + cur * 256;
#pragma unroll
        for (int r = 0; r < 8; ++r) {
            int m = r + mg;
            float s = acc[r] * SCALE_QK - 0.5f * (bq_r[r] + bk_n)
                    + affs * st[m * 16 + (lane & 15)];
            S[m * N_TOK + ncol] = s;
        }
        cur ^= 1;
    }
    __syncthreads();

    // ---- phase 2+3: per-row max, exact top-256 threshold (radix descent), masked exp ----
    for (int rr = wave; rr < 16; rr += 4) {
        float* Srow = &S[rr * N_TOK];
        float fm = -3.4e38f;
        for (int j = lane; j < N_TOK; j += 32) fm = fmaxf(fm, Srow[j]);
        for (int o = 16; o > 0; o >>= 1) fm = fmaxf(fm, __shfl_xor(fm, o, 32));

        unsigned P = 0u;                      // largest X with count(u >= X) >= K
        for (int b = 31; b >= 0; --b) {
            unsigned X = P | (1u << b);
            int cnt = 0;
            for (int j = lane; j < N_TOK; j += 32)
                cnt += (mono_u32(Srow[j]) >= X) ? 1 : 0;
            for (int o = 16; o > 0; o >>= 1) cnt += __shfl_xor(cnt, o, 32);
            if (cnt >= TOPK_K) P = X;
        }

        float lsum = 0.f;
        for (int j = lane; j < N_TOK; j += 32) {
            float s = Srow[j];
            float w = (mono_u32(s) >= P) ? __expf(s - fm) : 0.f;
            Srow[j] = w;
            lsum += w;
        }
        for (int o = 16; o > 0; o >>= 1) lsum += __shfl_xor(lsum, o, 32);
        if (lane == 0) { rowmax[rr] = fm; tauu[rr] = P; rowsum[rr] = lsum; }
    }
    __syncthreads();

    // ---- phase 4: out(16x48) = W(16x4096) x V(4096x48); wave owns a k-quarter ----
    v8f acc[3];
    acc[0] = (v8f){}; acc[1] = (v8f){}; acc[2] = (v8f){};
    const vf8* sp = (const vf8*)&S[(lane & 15) * N_TOK];
    for (int kk = 0; kk < 32; ++kk) {
        int k0 = (wave * 32 + kk) * 32;
        int ia = (k0 + kb8) >> 3;
        vf8 lo = sp[ia], hi = sp[ia + 2];
        v16bf a;
#pragma unroll
        for (int e = 0; e < 16; ++e)
            a[e] = (__bf16)((e < 8) ? lo[e] : hi[e - 8]);
#pragma unroll
        for (int nt = 0; nt < 3; ++nt) {
            const v8bf* vp = (const v8bf*)(vt +
                (size_t)(h * DHEAD + nt * 16 + (lane & 15)) * N_TOK);
            int ib = (k0 + kb8) >> 3;
            v16bf b = SHUF16(vp[ib], vp[ib + 2]);
            acc[nt] = wmma_bf16(a, b, acc[nt]);
        }
    }
    __syncthreads();          // all waves done reading S before reuse as scratch
#pragma unroll
    for (int nt = 0; nt < 3; ++nt)
#pragma unroll
        for (int r = 0; r < 8; ++r)
            S[((wave * 3 + nt) << 8) + (lane << 3) + r] = acc[nt][r];
    __syncthreads();

    for (int idx = tid; idx < 768; idx += 128) {   // 3 tiles x 256
        int nt = idx >> 8, pos = idx & 255;
        int ls = pos >> 3, r = pos & 7;
        float v = 0.f;
        for (int w = 0; w < 4; ++w) v += S[((w * 3 + nt) << 8) + pos];
        int m = r + ((ls >> 4) << 3);
        int c = ls & 15;
        O[(qt * 16 + m) * TD + h * DHEAD + nt * 16 + c] = v / rowsum[m];
    }
}

// ---------------- output projection + transpose to (192, 4096) ----------------
__global__ void k_outproj(const float* __restrict__ O, const float* __restrict__ Wo,
                          const float* __restrict__ bo, float* __restrict__ out) {
    int lane = threadIdx.x & 31;
    int wave = threadIdx.x >> 5;
    int rt = blockIdx.x;
    int ct = blockIdx.y * 4 + wave;

    int mg   = (lane >> 4) << 3;
    int kb8  = mg;
    int kb16 = (lane >> 4) << 4;
    int rowg = rt * 16 + (lane & 15);
    int colg = ct * 16 + (lane & 15);

    const vf8* op = (const vf8*)(O + rowg * TD);
    v8f acc = {};
    for (int kt = 0; kt < 6; ++kt) {   // K=192
        int k0 = kt * 32;
        int ia = (k0 + kb8) >> 3;
        vf8 lo = op[ia], hi = op[ia + 2];
        v16bf a, b;
#pragma unroll
        for (int e = 0; e < 16; ++e) {
            a[e] = (__bf16)((e < 8) ? lo[e] : hi[e - 8]);
            b[e] = (__bf16)Wo[(k0 + kb8 + (e & 7) + ((e >= 8) ? 16 : 0)) * TD + colg];
        }
        acc = wmma_bf16(a, b, acc);
    }
#pragma unroll
    for (int r = 0; r < 8; ++r) {
        int m = r + mg;
        out[colg * N_TOK + rt * 16 + m] = acc[r] + bo[colg];
    }
}

// ---------------- launch ----------------
extern "C" void kernel_launch(void* const* d_in, const int* in_sizes, int n_in,
                              void* d_out, int out_size, void* d_ws, size_t ws_size,
                              hipStream_t stream) {
    (void)in_sizes; (void)n_in; (void)out_size; (void)ws_size;
    const float* feat1  = (const float*)d_in[0];
    const float* feat2  = (const float*)d_in[1];
    const float* bmap   = (const float*)d_in[2];
    const float* segaff = (const float*)d_in[3];
    const float* Wq = (const float*)d_in[4];  const float* bq = (const float*)d_in[5];
    const float* Wk = (const float*)d_in[6];  const float* bk = (const float*)d_in[7];
    const float* Wv = (const float*)d_in[8];  const float* bv = (const float*)d_in[9];
    const float* Wo = (const float*)d_in[10]; const float* bo = (const float*)d_in[11];
    const float* c1w = (const float*)d_in[12]; const float* c1b = (const float*)d_in[13];
    const float* c2w = (const float*)d_in[14]; const float* c2b = (const float*)d_in[15];
    const float* affscale = (const float*)d_in[16];

    char* ws = (char*)d_ws;
    __bf16* qb   = (__bf16*)(ws + 0);         // 4*4096*64 bf16 = 2 MB
    __bf16* kbuf = (__bf16*)(ws + 2097152);   // 2 MB
    __bf16* vt   = (__bf16*)(ws + 4194304);   // V^T (4,48,4096) bf16 = 1.5 MB
    float*  bfl  = (float*) (ws + 5767168);   // 4*4096 f32
    float*  Obuf = (float*) (ws + 5832704);   // 4096*192 f32 = 3 MB

    // zero Q/K padded buffers (cols 48..63 per head stay zero)
    k_zero<<<1024, 256, 0, stream>>>((unsigned int*)ws, 4194304 / 4);

    k_conv<<<16, 256, 0, stream>>>(bmap, c1w, c1b, c2w, c2b, bfl);

    k_qkv<<<dim3(256, 3, 3), 128, 0, stream>>>(feat1, feat2, Wq, bq, Wk, bk, Wv, bv,
                                               qb, kbuf, vt);

    size_t lds_bytes = (size_t)(16 * N_TOK + 48) * sizeof(float); // 256 KB + tags
    k_attn<<<dim3(256, 4), 128, lds_bytes, stream>>>(qb, kbuf, vt, bfl, segaff,
                                                     affscale, Obuf);

    k_outproj<<<dim3(256, 3), 128, 0, stream>>>(Obuf, Wo, bo, (float*)d_out);
}
